// GATMultihead_4810363372831
// MI455X (gfx1250) — compile-verified, matrix operand order checked by reading, not run on hardware
//
#include <hip/hip_runtime.h>
#include <hip/hip_bf16.h>

// ---------------------------------------------------------------------------
// GAT multihead, CDNA5 (gfx1250), wave32, bf16 WMMA 16x16x32 with f32 accum.
// ---------------------------------------------------------------------------

#define NEGC   (-9000000000000000.0f)
#define SLOPEC (0.1f)

#define B_      4
#define N_      2048
#define NFEAT_  256
#define NHID_   64
#define NHEADS_ 8
#define NOUT_   128

typedef __bf16 bf16raw;
typedef bf16raw v16bf __attribute__((ext_vector_type(16)));
typedef float   v8f   __attribute__((ext_vector_type(8)));

union Frag {
    uint4          u4[2];
    unsigned short s[16];
    v16bf          v;
};

__device__ __forceinline__ unsigned short f2bf(float f) {
    unsigned int u = __float_as_uint(f);
    unsigned int r = (u + 0x7FFFu + ((u >> 16) & 1u)) >> 16;  // RNE
    return (unsigned short)r;
}
__device__ __forceinline__ float bf2f(unsigned short s) {
    return __uint_as_float(((unsigned int)s) << 16);
}
__device__ __forceinline__ void load8i(const int* __restrict__ p, int* d) {
    int4 a = ((const int4*)p)[0];
    int4 b = ((const int4*)p)[1];
    d[0]=a.x; d[1]=a.y; d[2]=a.z; d[3]=a.w;
    d[4]=b.x; d[5]=b.y; d[6]=b.z; d[7]=b.w;
}
__device__ __forceinline__ void load8f(const float* __restrict__ p, float* d) {
    float4 a = ((const float4*)p)[0];
    float4 b = ((const float4*)p)[1];
    d[0]=a.x; d[1]=a.y; d[2]=a.z; d[3]=a.w;
    d[4]=b.x; d[5]=b.y; d[6]=b.z; d[7]=b.w;
}

// ---------------------------------------------------------------------------
// Elementwise f32 -> bf16
// ---------------------------------------------------------------------------
__global__ void cvt_f32_to_bf16(const float* __restrict__ in,
                                unsigned short* __restrict__ out, int n) {
    int i = blockIdx.x * blockDim.x + threadIdx.x;
    if (i < n) out[i] = f2bf(in[i]);
}

// W (mat, K, OUT) f32 -> Wt (mat, OUT, K) bf16
__global__ void transpose_to_bf16(const float* __restrict__ W,
                                  unsigned short* __restrict__ Wt,
                                  int K, int OUT) {
    int mat = blockIdx.z;
    int idx = blockIdx.x * blockDim.x + threadIdx.x;
    if (idx >= K * OUT) return;
    int k = idx / OUT, o = idx % OUT;
    Wt[((size_t)mat * OUT + o) * K + k] = f2bf(W[((size_t)mat * K + k) * OUT + o]);
}

// ---------------------------------------------------------------------------
// Ht[mat][o][n] = sum_k X[batch][n][k] * Wt[head][o][k]   (bf16 in, bf16 out)
// one wave per 16-row x 64-col tile; K in chunks of 32 via wmma bf16.
// ---------------------------------------------------------------------------
__global__ void __launch_bounds__(32)
wmma_transform(const unsigned short* __restrict__ X,
               const unsigned short* __restrict__ Wt,
               unsigned short* __restrict__ Ht,
               int K, int N, int OUT, int headsPerBatch) {
    int tile_n = blockIdx.x;
    int tile_o = blockIdx.y;
    int mat    = blockIdx.z;
    int batch  = mat / headsPerBatch;
    int head   = mat % headsPerBatch;

    const unsigned short* Xm = X  + (size_t)batch * N * K;
    const unsigned short* Wm = Wt + (size_t)head * OUT * K;
    unsigned short*       Hm = Ht + (size_t)mat * OUT * N;

    int lane = threadIdx.x & 31;
    int lo   = lane & 15;
    int hf   = lane >> 4;  // 0: K{0..7,16..23}  1: K{8..15,24..31}
    int nrow = tile_n * 16 + lo;
    int h0   = tile_o * 64;

    v8f z = {0.f,0.f,0.f,0.f,0.f,0.f,0.f,0.f};
    v8f acc0 = z, acc1 = z, acc2 = z, acc3 = z;

    for (int k0 = 0; k0 < K; k0 += 32) {
        const unsigned short* xr = Xm + (size_t)nrow * K + k0 + hf * 8;
        Frag a;
        a.u4[0] = *(const uint4*)(xr);
        a.u4[1] = *(const uint4*)(xr + 16);

        Frag b0, b1, b2, b3;
        {
            const unsigned short* w0 = Wm + (size_t)(h0 +  0 + lo) * K + k0 + hf * 8;
            const unsigned short* w1 = Wm + (size_t)(h0 + 16 + lo) * K + k0 + hf * 8;
            const unsigned short* w2 = Wm + (size_t)(h0 + 32 + lo) * K + k0 + hf * 8;
            const unsigned short* w3 = Wm + (size_t)(h0 + 48 + lo) * K + k0 + hf * 8;
            b0.u4[0] = *(const uint4*)(w0); b0.u4[1] = *(const uint4*)(w0 + 16);
            b1.u4[0] = *(const uint4*)(w1); b1.u4[1] = *(const uint4*)(w1 + 16);
            b2.u4[0] = *(const uint4*)(w2); b2.u4[1] = *(const uint4*)(w2 + 16);
            b3.u4[0] = *(const uint4*)(w3); b3.u4[1] = *(const uint4*)(w3 + 16);
        }
        acc0 = __builtin_amdgcn_wmma_f32_16x16x32_bf16(false, a.v, false, b0.v, (short)0, acc0, false, false);
        acc1 = __builtin_amdgcn_wmma_f32_16x16x32_bf16(false, a.v, false, b1.v, (short)0, acc1, false, false);
        acc2 = __builtin_amdgcn_wmma_f32_16x16x32_bf16(false, a.v, false, b2.v, (short)0, acc2, false, false);
        acc3 = __builtin_amdgcn_wmma_f32_16x16x32_bf16(false, a.v, false, b3.v, (short)0, acc3, false, false);
    }

    // C layout: elem r -> row = r + hf*8, col = lane&15 (per 16-col tile)
    #pragma unroll
    for (int r = 0; r < 8; ++r) {
        int n = tile_n * 16 + r + hf * 8;
        Hm[(size_t)(h0 +  0 + lo) * N + n] = f2bf(acc0[r]);
        Hm[(size_t)(h0 + 16 + lo) * N + n] = f2bf(acc1[r]);
        Hm[(size_t)(h0 + 32 + lo) * N + n] = f2bf(acc2[r]);
        Hm[(size_t)(h0 + 48 + lo) * N + n] = f2bf(acc3[r]);
    }
}

// ---------------------------------------------------------------------------
// s1[n] = sum_f Ht[f][n]*a1[f] ; s2 likewise (coalesced over n).
// ---------------------------------------------------------------------------
__global__ void scores_kernel(const unsigned short* __restrict__ Ht,
                              const float* __restrict__ a,
                              float* __restrict__ S1, float* __restrict__ S2,
                              int OUT, int N, int headsPerBatch) {
    int mat  = blockIdx.y;
    int head = mat % headsPerBatch;
    int n    = blockIdx.x * blockDim.x + threadIdx.x;
    if (n >= N) return;
    const unsigned short* Hm = Ht + (size_t)mat * OUT * N;
    const float* a1 = a + (size_t)head * 2 * OUT;
    const float* a2 = a1 + OUT;
    float s1 = 0.f, s2 = 0.f;
    for (int f = 0; f < OUT; ++f) {
        float h = bf2f(Hm[(size_t)f * N + n]);
        s1 += h * a1[f];
        s2 += h * a2[f];
    }
    S1[(size_t)mat * N + n] = s1;
    S2[(size_t)mat * N + n] = s2;
}

// ---------------------------------------------------------------------------
// Flash-style masked attention + aggregation.  One wave = 16 output rows.
// TILES*16 = OUT (64 -> 4, 128 -> 8).  FINAL: relu->f32 out, else elu->bf16.
// ---------------------------------------------------------------------------
template <int TILES, bool FINAL>
__global__ void __launch_bounds__(32)
attn_kernel(const int* __restrict__ adj,
            const float* __restrict__ S1, const float* __restrict__ S2,
            const unsigned short* __restrict__ Ht,   // (mat, OUT, N) bf16
            const float* __restrict__ bias,          // (heads, OUT)
            unsigned short* __restrict__ dstB,       // (B, N, heads*OUT) bf16
            float* __restrict__ dstF,                // (B, N, OUT) f32
            int N, int headsPerBatch) {
    constexpr int OUT = TILES * 16;
    int tile_n = blockIdx.x;
    int mat    = blockIdx.y;
    int batch  = mat / headsPerBatch;
    int head   = mat % headsPerBatch;

    int lane = threadIdx.x & 31;
    int lo   = lane & 15;
    int hf   = lane >> 4;

    int row_g = tile_n * 16 + lo;  // this lane's attention row
    const int*   adjRow = adj + ((size_t)batch * N + row_g) * N;
    const float* S2m    = S2 + (size_t)mat * N;
    const unsigned short* Hm = Ht + (size_t)mat * OUT * N;
    float s1v = S1[(size_t)mat * N + row_g];

    v8f z = {0.f,0.f,0.f,0.f,0.f,0.f,0.f,0.f};
    v8f acc[TILES];
    #pragma unroll
    for (int t = 0; t < TILES; ++t) acc[t] = z;

    float mrow = -3.0e38f;
    float lrow = 0.f;

    for (int j0 = 0; j0 < N; j0 += 32) {
        int k0 = j0 + hf * 8;        // first 8 neighbor cols for this lane
        int k1 = j0 + 16 + hf * 8;   // second 8

        __builtin_prefetch((const void*)(adjRow + j0 + 64), 0, 1);

        int   av[16];
        float sv[16];
        load8i(adjRow + k0, av);
        load8i(adjRow + k1, av + 8);
        load8f(S2m + k0, sv);
        load8f(S2m + k1, sv + 8);

        float e[16];
        #pragma unroll
        for (int i = 0; i < 16; ++i) {
            float v = s1v + sv[i];
            v = (v >= 0.f) ? v : SLOPEC * v;
            e[i] = (av[i] > 0) ? v : NEGC;
        }

        float cm = e[0];
        #pragma unroll
        for (int i = 1; i < 16; ++i) cm = fmaxf(cm, e[i]);
        cm = fmaxf(cm, __shfl_xor(cm, 16, 32));   // pair lanes share a row
        float mnew  = fmaxf(mrow, cm);
        float alpha = __expf(mrow - mnew);

        float ps = 0.f;
        Frag P;
        #pragma unroll
        for (int i = 0; i < 16; ++i) {
            float p = __expf(e[i] - mnew);
            ps += p;
            P.s[i] = f2bf(p);
        }
        ps += __shfl_xor(ps, 16, 32);
        lrow = lrow * alpha + ps;
        mrow = mnew;

        // rescale accumulators: elem r belongs to row (r + hf*8); its alpha
        // lives in lane (r + hf*8).
        #pragma unroll
        for (int r = 0; r < 8; ++r) {
            float ar = __shfl(alpha, r + hf * 8, 32);
            #pragma unroll
            for (int t = 0; t < TILES; ++t) acc[t][r] *= ar;
        }

        // B fragments: col = t*16 + lo (output feature), K = neighbor index.
        #pragma unroll
        for (int t = 0; t < TILES; ++t) {
            const unsigned short* hb = Hm + (size_t)(t * 16 + lo) * N;
            Frag b;
            b.u4[0] = *(const uint4*)(hb + k0);
            b.u4[1] = *(const uint4*)(hb + k1);
            acc[t] = __builtin_amdgcn_wmma_f32_16x16x32_bf16(
                false, P.v, false, b.v, (short)0, acc[t], false, false);
        }
    }

    // Epilogue: normalize, bias, activation, store.
    #pragma unroll
    for (int r = 0; r < 8; ++r) {
        float linv = 1.f / __shfl(lrow, r + hf * 8, 32);
        int n = tile_n * 16 + r + hf * 8;
        #pragma unroll
        for (int t = 0; t < TILES; ++t) {
            int o = t * 16 + lo;
            float v = acc[t][r] * linv + bias[(size_t)head * OUT + o];
            if constexpr (FINAL) {
                v = fmaxf(v, 0.f);  // relu
                dstF[((size_t)batch * N + n) * OUT + o] = v;
            } else {
                v = (v > 0.f) ? v : (__expf(v) - 1.f);  // elu
                dstB[((size_t)batch * N + n) * (size_t)(headsPerBatch * OUT) +
                     (size_t)head * OUT + o] = f2bf(v);
            }
        }
    }
}

// ---------------------------------------------------------------------------
// Host launcher
// ---------------------------------------------------------------------------
extern "C" void kernel_launch(void* const* d_in, const int* in_sizes, int n_in,
                              void* d_out, int out_size, void* d_ws, size_t ws_size,
                              hipStream_t stream) {
    (void)in_sizes; (void)n_in; (void)out_size; (void)ws_size;

    const float* x   = (const float*)d_in[0];  // (B,N,NFEAT)
    const int*   adj = (const int*)  d_in[1];  // (B,N,N)
    const float* Wh  = (const float*)d_in[2];  // (NHEADS,NFEAT,NHID)
    const float* ah  = (const float*)d_in[3];  // (NHEADS,2*NHID)
    const float* bh  = (const float*)d_in[4];  // (NHEADS,NHID)
    const float* Wo  = (const float*)d_in[5];  // (NHID*NHEADS,NOUT)
    const float* ao  = (const float*)d_in[6];  // (2*NOUT)
    const float* bo  = (const float*)d_in[7];  // (NOUT)
    float*       out = (float*)d_out;          // (B,N,NOUT)

    // Workspace carve-up (256B aligned)
    size_t off = 0;
    auto carve = [&](size_t bytes) -> void* {
        void* p = (char*)d_ws + off;
        off += (bytes + 255) & ~(size_t)255;
        return p;
    };
    unsigned short* xb   = (unsigned short*)carve((size_t)B_ * N_ * NFEAT_ * 2);
    unsigned short* Wht  = (unsigned short*)carve((size_t)NHEADS_ * NHID_ * NFEAT_ * 2);
    unsigned short* Ht1  = (unsigned short*)carve((size_t)B_ * NHEADS_ * NHID_ * N_ * 2);
    float*          S1a  = (float*)carve((size_t)B_ * NHEADS_ * N_ * 4);
    float*          S2a  = (float*)carve((size_t)B_ * NHEADS_ * N_ * 4);
    unsigned short* hcat = (unsigned short*)carve((size_t)B_ * N_ * NHEADS_ * NHID_ * 2);
    unsigned short* Wot  = (unsigned short*)carve((size_t)NOUT_ * NHEADS_ * NHID_ * 2);
    unsigned short* Ht2  = (unsigned short*)carve((size_t)B_ * NOUT_ * N_ * 2);
    float*          S1b  = (float*)carve((size_t)B_ * N_ * 4);
    float*          S2b  = (float*)carve((size_t)B_ * N_ * 4);

    const int K2 = NHEADS_ * NHID_;  // 512

    // --- precision conversion / weight transposes ---
    {
        int n = B_ * N_ * NFEAT_;
        cvt_f32_to_bf16<<<dim3((n + 255) / 256), dim3(256), 0, stream>>>(x, xb, n);
    }
    transpose_to_bf16<<<dim3((NFEAT_ * NHID_ + 255) / 256, 1, NHEADS_),
                        dim3(256), 0, stream>>>(Wh, Wht, NFEAT_, NHID_);
    transpose_to_bf16<<<dim3((K2 * NOUT_ + 255) / 256, 1, 1),
                        dim3(256), 0, stream>>>(Wo, Wot, K2, NOUT_);

    // --- layer 1: transform, scores, attention+elu ---
    wmma_transform<<<dim3(N_ / 16, NHID_ / 64, B_ * NHEADS_), dim3(32), 0, stream>>>(
        xb, Wht, Ht1, NFEAT_, N_, NHID_, NHEADS_);
    scores_kernel<<<dim3(N_ / 256, B_ * NHEADS_), dim3(256), 0, stream>>>(
        Ht1, ah, S1a, S2a, NHID_, N_, NHEADS_);
    attn_kernel<4, false><<<dim3(N_ / 16, B_ * NHEADS_), dim3(32), 0, stream>>>(
        adj, S1a, S2a, Ht1, bh, hcat, nullptr, N_, NHEADS_);

    // --- layer 2: transform, scores, attention+relu -> d_out ---
    wmma_transform<<<dim3(N_ / 16, NOUT_ / 64, B_), dim3(32), 0, stream>>>(
        hcat, Wot, Ht2, K2, N_, NOUT_, 1);
    scores_kernel<<<dim3(N_ / 256, B_), dim3(256), 0, stream>>>(
        Ht2, ao, S1b, S2b, NOUT_, N_, 1);
    attn_kernel<8, true><<<dim3(N_ / 16, B_), dim3(32), 0, stream>>>(
        adj, S1b, S2b, Ht2, bo, nullptr, out, N_, 1);
}
